// LinearAttention_82755429859747
// MI455X (gfx1250) — matrix operand order repaired and checked by reading, not compiled, not run on hardware
//
#include <hip/hip_runtime.h>

typedef _Float16 h16;
typedef __attribute__((ext_vector_type(16))) _Float16 v16h;
typedef __attribute__((ext_vector_type(8)))  _Float16 v8h;
typedef __attribute__((ext_vector_type(8)))  float    v8f;

static constexpr int BB = 4, LQ = 4096, LK = 4096;
static constexpr int DQ = 256, DK = 64, DV = 128, DO = 256;

// ---------- WMMA helpers (CDNA5 wave32 layouts, CDNA5 ISA) ----------

__device__ __forceinline__ v8f wmma16(v16h a, v16h b, v8f c) {
  // D = A(16x32 f16) * B(32x16 f16) + C(16x16 f32)
  return __builtin_amdgcn_wmma_f32_16x16x32_f16(false, a, false, b, (short)0, c,
                                                false, false);
}

// A-tile 16x32 from f16 row-major [row][k] (ld halfs). Doc layout:
// lane<16: K={half*8..+7} U {16+half*8..+7}; per-lane two contiguous 16B loads.
__device__ __forceinline__ v16h load_a16(const h16* base, int ld, int lane) {
  int r = lane & 15, half = lane >> 4;
  const h16* p = base + r * ld + half * 8;
  union { v16h v; v8h q[2]; } u;
  u.q[0] = *(const v8h*)(p);
  u.q[1] = *(const v8h*)(p + 16);
  return u.v;
}

// Same A-tile but from fp32 row-major, converting to f16 in-register.
__device__ __forceinline__ v16h load_a32(const float* base, int ld, int lane) {
  int r = lane & 15, half = lane >> 4;
  const float* p = base + r * ld + half * 8;
  v16h a;
#pragma unroll
  for (int e = 0; e < 8; ++e) {
    a[e]     = (_Float16)p[e];
    a[e + 8] = (_Float16)p[16 + e];
  }
  return a;
}

// B-tile 32x16 from *transposed* storage Bt[n][k] (ld halfs): lane n = col,
// K = {half*8..+7} U {16+half*8..+7} (matches A) -> two 16B loads per lane.
__device__ __forceinline__ v16h load_b16(const h16* bt, int ld, int lane) {
  int n = lane & 15, half = lane >> 4;
  const h16* p = bt + n * ld + half * 8;
  union { v16h v; v8h q[2]; } u;
  u.q[0] = *(const v8h*)(p);
  u.q[1] = *(const v8h*)(p + 16);
  return u.v;
}

// ---------- async global->LDS (GLOBAL_LOAD_ASYNC_TO_LDS_B128, ASYNCcnt) ----------

__device__ __forceinline__ void async_g2l_16B(h16* lds, const h16* g) {
  unsigned l = (unsigned)(uintptr_t)lds;  // low 32b of generic ptr = LDS offset
  asm volatile("global_load_async_to_lds_b128 %0, %1, off"
               :: "v"(l), "v"(g) : "memory");
}
__device__ __forceinline__ void wait_async() {
  asm volatile("s_wait_asynccnt 0" ::: "memory");
}

// ---------- weight transpose + fp32->f16 convert: Wt[n*K + k] = W[k*Nc + n] ----------

__global__ void la_wtrans(const float* __restrict__ W, h16* __restrict__ Wt,
                          int K, int Nc) {
  int i = blockIdx.x * blockDim.x + threadIdx.x;
  if (i >= K * Nc) return;
  int k = i % K, n = i / K;
  Wt[i] = (h16)W[(size_t)k * Nc + n];
}

// ---------- projection GEMM: out = f16(X[rows,KD] @ W + b) via WMMA ----------

template <int KD, int NC, bool TR>
__global__ __launch_bounds__(256) void la_proj(const float* __restrict__ X,
                                               const h16* __restrict__ Wt,
                                               const float* __restrict__ bias,
                                               h16* __restrict__ out, int Ntot) {
  int lane = threadIdx.x & 31, w = threadIdx.x >> 5;
  int l15 = lane & 15, half = lane >> 4;
  int r0 = blockIdx.x * 128 + w * 16;  // 8 waves x 16 rows
  const float* Xp = X + (size_t)r0 * KD;

  v16h areg[KD / 32];
#pragma unroll
  for (int s = 0; s < KD / 32; ++s) areg[s] = load_a32(Xp + s * 32, KD, lane);

#pragma unroll
  for (int nt = 0; nt < NC / 16; ++nt) {
    int c0 = nt * 16;
    float bv = bias[c0 + l15];
    v8f acc;
#pragma unroll
    for (int v = 0; v < 8; ++v) acc[v] = bv;  // bias-initialized C
#pragma unroll
    for (int s = 0; s < KD / 32; ++s) {
      v16h bm = load_b16(Wt + (size_t)c0 * KD + s * 32, KD, lane);
      acc = wmma16(areg[s], bm, acc);
    }
    if (TR) {  // 8 consecutive rows per lane -> one 16B store
      v8h pk;
#pragma unroll
      for (int v = 0; v < 8; ++v) pk[v] = (h16)acc[v];
      *(v8h*)(out + (size_t)(c0 + l15) * Ntot + r0 + half * 8) = pk;
    } else {
#pragma unroll
      for (int v = 0; v < 8; ++v)
        out[(size_t)(r0 + v + half * 8) * NC + c0 + l15] = (h16)acc[v];
    }
  }
}

// ---------- fused streaming attention + output projection ----------
// block: 64 q-rows, 8 wave32s, TK=128 chunks, double-buffered async K/V staging.
// Qh,Kh row-major f16 [.][64]; Vt f16 [c*B + b][LK]; Wot f16 [DO][128].

static constexpr int TK = 128;

__global__ __launch_bounds__(256) void la_attn(const h16* __restrict__ Qh,
                                               const h16* __restrict__ Kh,
                                               const h16* __restrict__ Vt,
                                               const h16* __restrict__ Wot,
                                               const float* __restrict__ bo,
                                               float* __restrict__ out) {
  // carved LDS (112 KB): sK[2] 2x16KB | sV[2] 2x32KB | sS 16KB ; sO aliases sK.
  alignas(128) __shared__ h16 smem[2 * TK * 64 + 2 * 128 * TK + 64 * TK];
  h16* sK0 = smem;                    // [TK][64]
  h16* sK1 = smem + TK * 64;
  h16* sV0 = smem + 2 * TK * 64;      // [128][TK] (transposed V)
  h16* sV1 = sV0 + 128 * TK;
  h16* sS  = sV1 + 128 * TK;          // [64][TK]
  h16* sO  = smem;                    // [64][128], reuses sK after main loop

  int tid = threadIdx.x, lane = tid & 31, w = tid >> 5;
  int l15 = lane & 15, half = lane >> 4;
  int b = blockIdx.y, q0 = blockIdx.x * 64;

  // wave mapping: mw in [0,4) -> 16 rows; nw selects 64 O-cols / 64 S-cols
  int mw = w >> 1, nw = w & 1;

  // per-wave Q operand, loaded once, straight from global in A layout
  const h16* Qg = Qh + ((size_t)(b * LQ + q0 + mw * 16)) * 64;
  v16h qa0 = load_a16(Qg, 64, lane);
  v16h qa1 = load_a16(Qg + 32, 64, lane);

  const h16* Kg = Kh + (size_t)(b * LK) * 64;
  int vcol = tid >> 1, vj = (tid & 1) * 64;
  const h16* Vg = Vt + ((size_t)vcol * BB + b) * LK;

  auto stage = [&](int kk, h16* kbuf, h16* vbuf) {  // 16KB K + 32KB V, ASYNCcnt
    const h16* kg = Kg + (size_t)kk * 64 + tid * 32;
    h16* kd = kbuf + tid * 32;
#pragma unroll
    for (int u = 0; u < 4; ++u) async_g2l_16B(kd + u * 8, kg + u * 8);
    const h16* vg = Vg + kk + vj;
    h16* vd = vbuf + vcol * TK + vj;
#pragma unroll
    for (int u = 0; u < 8; ++u) async_g2l_16B(vd + u * 8, vg + u * 8);
  };

  stage(0, sK0, sV0);
  wait_async();
  __syncthreads();

  v8f acc[4];
#pragma unroll
  for (int t = 0; t < 4; ++t)
#pragma unroll
    for (int v = 0; v < 8; ++v) acc[t][v] = 0.0f;

  for (int kk = 0; kk < LK; kk += TK) {
    int ph = (kk / TK) & 1;
    h16* sKp = ph ? sK1 : sK0;
    h16* sVp = ph ? sV1 : sV0;
    if (kk + TK < LK) stage(kk + TK, ph ? sK0 : sK1, ph ? sV0 : sV1);

    // ---- S = Q * K_chunk^T : wave does 4 tiles (rows mw*16, cols nw*64..+63)
    // double-buffered S accumulator: converts of tile j overlap WMMAs of j+1
    {
      int c0 = nw * 64;
      v8f scur;
#pragma unroll
      for (int v = 0; v < 8; ++v) scur[v] = 0.0f;
      scur = wmma16(qa0, load_b16(sKp + c0 * 64, 64, lane), scur);
      scur = wmma16(qa1, load_b16(sKp + c0 * 64 + 32, 64, lane), scur);
#pragma unroll
      for (int j = 0; j < 4; ++j) {
        v8f snext;
        if (j < 3) {
          int c1 = nw * 64 + (j + 1) * 16;
#pragma unroll
          for (int v = 0; v < 8; ++v) snext[v] = 0.0f;
          snext = wmma16(qa0, load_b16(sKp + c1 * 64, 64, lane), snext);
          snext = wmma16(qa1, load_b16(sKp + c1 * 64 + 32, 64, lane), snext);
        }
        int cj = nw * 64 + j * 16;
#pragma unroll
        for (int v = 0; v < 8; ++v)
          sS[(mw * 16 + v + half * 8) * TK + cj + l15] = (h16)scur[v];
        if (j < 3) scur = snext;
      }
    }
    __syncthreads();  // publish S

    // ---- O += S * V_chunk : 4 O-tiles/wave, inner dim 128 (4 slices)
    v16h a[4];
#pragma unroll
    for (int s = 0; s < 4; ++s)
      a[s] = load_a16(sS + mw * 16 * TK + s * 32, TK, lane);
    {
      const h16* vb = sVp + (nw * 64) * TK;
      v16h bcur = load_b16(vb, TK, lane);  // (t=0,s=0)
#pragma unroll
      for (int t = 0; t < 4; ++t) {
#pragma unroll
        for (int s = 0; s < 4; ++s) {
          v16h bnext;
          if (!(t == 3 && s == 3)) {  // prefetch next B one WMMA ahead
            int tn = (s == 3) ? t + 1 : t;
            int sn = (s == 3) ? 0 : s + 1;
            bnext = load_b16(sVp + (nw * 64 + tn * 16) * TK + sn * 32, TK, lane);
          }
          acc[t] = wmma16(a[s], bcur, acc[t]);
          if (!(t == 3 && s == 3)) bcur = bnext;
        }
      }
    }

    wait_async();      // next chunk resident in LDS
    __syncthreads();   // readers done with current buffers; staged buf visible
  }

  // spill O to LDS (aliases sK region; all sK reads are behind the last barrier)
#pragma unroll
  for (int t = 0; t < 4; ++t)
#pragma unroll
    for (int v = 0; v < 8; ++v)
      sO[(mw * 16 + v + half * 8) * 128 + nw * 64 + t * 16 + l15] = (h16)acc[t][v];
  __syncthreads();

  // out = O @ Wo + bo : wave -> (m-tile = w>>1, 8 of 16 N-tiles)
  int fmt = w >> 1;
  v16h afin[4];
#pragma unroll
  for (int s = 0; s < 4; ++s)
    afin[s] = load_a16(sO + fmt * 16 * 128 + s * 32, 128, lane);

  for (int j = 0; j < 8; ++j) {
    int c0 = ((w & 1) * 8 + j) * 16;
    float bv = bo[c0 + l15];
    v8f oc;
#pragma unroll
    for (int v = 0; v < 8; ++v) oc[v] = bv;
#pragma unroll
    for (int s = 0; s < 4; ++s) {
      v16h bm = load_b16(Wot + (size_t)c0 * 128 + s * 32, 128, lane);  // L2-resident
      oc = wmma16(afin[s], bm, oc);
    }
#pragma unroll
    for (int v = 0; v < 8; ++v) {
      size_t row = (size_t)b * LQ + q0 + fmt * 16 + v + half * 8;
      out[row * DO + c0 + l15] = oc[v];
    }
  }
}

// ---------- host launch ----------

extern "C" void kernel_launch(void* const* d_in, const int* in_sizes, int n_in,
                              void* d_out, int out_size, void* d_ws, size_t ws_size,
                              hipStream_t stream) {
  const float* query = (const float*)d_in[0];
  const float* key   = (const float*)d_in[1];
  const float* value = (const float*)d_in[2];
  const float* Wq    = (const float*)d_in[3];
  const float* bq    = (const float*)d_in[4];
  const float* Wk    = (const float*)d_in[5];
  const float* bk    = (const float*)d_in[6];
  const float* Wv    = (const float*)d_in[7];
  const float* bv    = (const float*)d_in[8];
  const float* Wo    = (const float*)d_in[9];
  const float* bo    = (const float*)d_in[10];
  float* out = (float*)d_out;

  uint8_t* ws = (uint8_t*)d_ws;
  size_t off = 0;
  auto carve = [&](size_t bytes) -> void* {
    void* p = ws + off;
    off = (off + bytes + 255) & ~(size_t)255;
    return p;
  };
  const size_t NQ = (size_t)BB * LQ, NK = (size_t)BB * LK;
  h16* Qh  = (h16*)carve(NQ * DK * 2);        // [B*LQ][64]
  h16* Kh  = (h16*)carve(NK * DK * 2);        // [B*LK][64]
  h16* Vt  = (h16*)carve(NK * DV * 2);        // [c*B + b][LK]
  h16* Wqt = (h16*)carve((size_t)DK * DQ * 2);
  h16* Wkt = (h16*)carve((size_t)DK * DK * 2);
  h16* Wvt = (h16*)carve((size_t)DV * DV * 2);
  h16* Wot = (h16*)carve((size_t)DO * DV * 2);

  auto g1 = [](int n) { return dim3((n + 255) / 256); };
  la_wtrans<<<g1(DQ * DK), 256, 0, stream>>>(Wq, Wqt, DQ, DK);
  la_wtrans<<<g1(DK * DK), 256, 0, stream>>>(Wk, Wkt, DK, DK);
  la_wtrans<<<g1(DV * DV), 256, 0, stream>>>(Wv, Wvt, DV, DV);
  la_wtrans<<<g1(DV * DO), 256, 0, stream>>>(Wo, Wot, DV, DO);

  la_proj<DQ, DK, false><<<dim3((int)(NQ / 128)), 256, 0, stream>>>(
      query, Wqt, bq, Qh, (int)NQ);
  la_proj<DK, DK, false><<<dim3((int)(NK / 128)), 256, 0, stream>>>(
      key, Wkt, bk, Kh, (int)NK);
  la_proj<DV, DV, true><<<dim3((int)(NK / 128)), 256, 0, stream>>>(
      value, Wvt, bv, Vt, (int)NK);

  la_attn<<<dim3(LQ / 64, BB), 256, 0, stream>>>(Qh, Kh, Vt, Wot, bo, out);
}
